// _AWQ4Linear_39024072852004
// MI455X (gfx1250) — matrix operand order repaired and checked by reading, not compiled, not run on hardware
//
#include <hip/hip_runtime.h>

// ---------------------------------------------------------------------------
// AWQ int4 linear as a bf16 WMMA GEMM for gfx1250 (MI455X).
// out[m,o] = sum_k x[m,k]*col_scale[k] * (s[o,g]*q[o,k] + z[o,g]) + bias[o]
// M=8192, K=4096, N=11008.  Block tile 256x128x32, 8 waves, wave tile 64x64.
// Dequant trick: bf16(0x4380|q) == 256+2q exactly, so
//   s*q + z == (s/2)*bf16(0x4380|q) + (z-128*s)   -> one v_pk_fma_bf16 per byte.
// Register budget: acc 128 + bfrag 32 + afrag 8 (JIT-loaded) + staging ~24
// keeps peak allocation under the 256-VGPR MSB boundary.
// ---------------------------------------------------------------------------

#define IN_F    4096
#define OUT_F   11008
#define M_ROWS  8192
#define NGROUPS 64
#define WQ_COLS 2048
#define BM      256
#define BN      128
#define BK      32
#define NK      (IN_F / BK)   // 128
#define LDSK    40            // padded row stride (bf16): 80B, 16B-aligned, conflict-free

typedef __attribute__((ext_vector_type(16))) __bf16 v16bf;
typedef __attribute__((ext_vector_type(8)))  __bf16 v8bf;
typedef __attribute__((ext_vector_type(2)))  __bf16 v2bf;
typedef __attribute__((ext_vector_type(8)))  float  v8f;

__global__ __launch_bounds__(256)
void awq4_wmma_gemm(const float* __restrict__ x,
                    const int*   __restrict__ wq,
                    const float* __restrict__ scales,
                    const float* __restrict__ zeros,
                    const float* __restrict__ cs,
                    const float* __restrict__ bias,
                    float*       __restrict__ out)
{
    __shared__ __bf16 As[2][BM * LDSK];   // 2 x 20480 B
    __shared__ __bf16 Bs[2][BN * LDSK];   // 2 x 10240 B
    __shared__ float  csl[IN_F];          // 16384 B  (col_scale, staged once)

    const int tid  = threadIdx.x;
    const int lane = tid & 31;
    const int wav  = tid >> 5;            // 8 waves
    const int wm   = wav & 3;             // 4 wave-rows of 64
    const int wn   = wav >> 2;            // 2 wave-cols of 64
    const int lq   = lane & 15;
    const int lh   = lane >> 4;

    const int mblock = blockIdx.y * BM;
    const int nblock = blockIdx.x * BN;

    // A staging: 4 lanes per row (8 K each), 4 row-passes of 64
    const int ak = (tid & 3) * 8;         // K offset within tile
    const int ar = tid >> 2;              // row 0..63 (+64*pass)
    // B staging: 2 threads per out-feature row, 16 K each
    const int brow = tid >> 1, bhalf = tid & 1;
    const int oc   = nblock + brow;

    // ---- stage col_scale to LDS once -----------------------------------
    #pragma unroll
    for (int j = 0; j < 4; ++j) {
        const int idx = tid * 16 + j * 4;           // covers 4096 floats
        *(float4*)&csl[idx] = *(const float4*)&cs[idx];
    }

    v8f acc[4][4];
    #pragma unroll
    for (int tm = 0; tm < 4; ++tm)
        #pragma unroll
        for (int tn = 0; tn < 4; ++tn)
            acc[tm][tn] = (v8f){0.f,0.f,0.f,0.f,0.f,0.f,0.f,0.f};

    // ---- register-stage loaders ----------------------------------------
    auto load_a = [&](int kt, v8bf (&sa)[4]) {
        float4 c0 = *(const float4*)&csl[kt * BK + ak];
        float4 c1 = *(const float4*)&csl[kt * BK + ak + 4];
        float ca[8] = {c0.x, c0.y, c0.z, c0.w, c1.x, c1.y, c1.z, c1.w};
        #pragma unroll
        for (int p = 0; p < 4; ++p) {
            const float* xp = x + (size_t)(mblock + p * 64 + ar) * IN_F + kt * BK + ak;
            float4 x0 = *(const float4*)(xp);
            float4 x1 = *(const float4*)(xp + 4);
            float xa[8] = {x0.x, x0.y, x0.z, x0.w, x1.x, x1.y, x1.z, x1.w};
            #pragma unroll
            for (int i = 0; i < 8; ++i)
                sa[p][i] = (__bf16)(xa[i] * ca[i]);   // v_fma_mix / cvt_pk_bf16
        }
    };

    auto load_b = [&](int kt, v8bf& b0, v8bf& b1) {
        const int* wp = wq + (size_t)oc * WQ_COLS + kt * 16 + bhalf * 8;
        int4 q0 = *(const int4*)(wp);
        int4 q1 = *(const int4*)(wp + 4);
        const int g = kt >> 1;
        const float s = scales[oc * NGROUPS + g];
        const float z = zeros [oc * NGROUPS + g];
        const float sh = 0.5f * s;
        const float zh = __builtin_fmaf(-128.f, s, z);
        v2bf s2; s2[0] = (__bf16)sh; s2[1] = (__bf16)sh;
        v2bf z2; z2[0] = (__bf16)zh; z2[1] = (__bf16)zh;
        const unsigned qq[8] = {(unsigned)q0.x, (unsigned)q0.y, (unsigned)q0.z, (unsigned)q0.w,
                                (unsigned)q1.x, (unsigned)q1.y, (unsigned)q1.z, (unsigned)q1.w};
        uint4 r0, r1;
        unsigned* p0 = &r0.x; unsigned* p1 = &r1.x;
        #pragma unroll
        for (int j = 0; j < 4; ++j) {
            // packed bf16 pair: lo = 0x4380|lo_nib (==256+2q_lo), hi likewise
            unsigned t0 = 0x43804380u | (qq[j]     & 0xFu) | ((qq[j]     << 12) & 0x000F0000u);
            unsigned t1 = 0x43804380u | (qq[j + 4] & 0xFu) | ((qq[j + 4] << 12) & 0x000F0000u);
            v2bf d0 = __builtin_bit_cast(v2bf, t0) * s2 + z2;   // v_pk_fma_bf16
            v2bf d1 = __builtin_bit_cast(v2bf, t1) * s2 + z2;
            p0[j] = __builtin_bit_cast(unsigned, d0);
            p1[j] = __builtin_bit_cast(unsigned, d1);
        }
        b0 = __builtin_bit_cast(v8bf, r0);
        b1 = __builtin_bit_cast(v8bf, r1);
    };

    auto store_tile = [&](int buf, const v8bf (&sa)[4], v8bf b0, v8bf b1) {
        #pragma unroll
        for (int p = 0; p < 4; ++p)
            *(v8bf*)&As[buf][(p * 64 + ar) * LDSK + ak] = sa[p];
        __bf16* pb = &Bs[buf][brow * LDSK + bhalf * 16];
        *(v8bf*)(pb)     = b0;
        *(v8bf*)(pb + 8) = b1;
    };

    // ---- WMMA compute from one LDS buffer ------------------------------
    // B fragments are reused by all 4 tm steps -> keep resident.
    // A fragments are used once each -> load just-in-time (1 live, not 4).
    auto compute = [&](int buf) {
        v16bf bfrag[4];
        #pragma unroll
        for (int tn = 0; tn < 4; ++tn) {
            const __bf16* p = &Bs[buf][(wn * 64 + tn * 16 + lq) * LDSK + lh * 16];
            v8bf lo = *(const v8bf*)(p);
            v8bf hi = *(const v8bf*)(p + 8);
            bfrag[tn] = __builtin_shufflevector(lo, hi,
                0,1,2,3,4,5,6,7,8,9,10,11,12,13,14,15);
        }
        #pragma unroll
        for (int tm = 0; tm < 4; ++tm) {
            const __bf16* p = &As[buf][(wm * 64 + tm * 16 + lq) * LDSK + lh * 16];
            v8bf lo = *(const v8bf*)(p);
            v8bf hi = *(const v8bf*)(p + 8);
            v16bf af = __builtin_shufflevector(lo, hi,
                0,1,2,3,4,5,6,7,8,9,10,11,12,13,14,15);
            #pragma unroll
            for (int tn = 0; tn < 4; ++tn)
                acc[tm][tn] = __builtin_amdgcn_wmma_f32_16x16x32_bf16(
                    false, af, false, bfrag[tn],
                    (short)0, acc[tm][tn], false, false);
        }
    };

    __syncthreads();                       // csl visible before first load_a

    // ---- software-pipelined main loop (double-buffered LDS) ------------
    {
        v8bf sa[4], b0, b1;
        load_a(0, sa);
        load_b(0, b0, b1);
        store_tile(0, sa, b0, b1);
    }
    __syncthreads();

    int buf = 0;
    for (int kt = 0; kt < NK; ++kt) {
        const bool more = (kt + 1) < NK;
        v8bf sa[4], b0, b1;
        if (more) {
            load_a(kt + 1, sa);
            load_b(kt + 1, b0, b1);
        }
        if (kt + 2 < NK) {
            __builtin_prefetch(wq + (size_t)oc * WQ_COLS + (kt + 2) * 16, 0, 1);
            __builtin_prefetch(x + (size_t)(mblock + ar) * IN_F + (kt + 2) * BK, 0, 1);
        }
        compute(buf);
        if (more) store_tile(buf ^ 1, sa, b0, b1);
        __syncthreads();
        buf ^= 1;
    }

    // ---- epilogue: bias + store ----------------------------------------
    #pragma unroll
    for (int tn = 0; tn < 4; ++tn) {
        const int n   = nblock + wn * 64 + tn * 16 + lq;
        const float bv = bias[n];
        #pragma unroll
        for (int tm = 0; tm < 4; ++tm) {
            const int mb = mblock + wm * 64 + tm * 16 + lh * 8;
            #pragma unroll
            for (int v = 0; v < 8; ++v)
                out[(size_t)(mb + v) * OUT_F + n] = acc[tm][tn][v] + bv;
        }
    }
}

extern "C" void kernel_launch(void* const* d_in, const int* in_sizes, int n_in,
                              void* d_out, int out_size, void* d_ws, size_t ws_size,
                              hipStream_t stream) {
    const float* x      = (const float*)d_in[0];
    const int*   wq     = (const int*)  d_in[1];
    const float* scales = (const float*)d_in[2];
    const float* zeros  = (const float*)d_in[3];
    const float* cs     = (const float*)d_in[4];
    const float* bias   = (const float*)d_in[5];
    float*       out    = (float*)d_out;

    dim3 grid(OUT_F / BN, M_ROWS / BM);   // (86, 32)
    awq4_wmma_gemm<<<grid, 256, 0, stream>>>(x, wq, scales, zeros, cs, bias, out);
}
